// GPTAdapter_73143293051205
// MI455X (gfx1250) — matrix operand order repaired and checked by reading, not compiled, not run on hardware
//
#include <hip/hip_runtime.h>
#include <hip/hip_bf16.h>
#include <math.h>

// ---------------------------------------------------------------------------
// GPT forward for gfx1250 (CDNA5). All GEMMs + attention run on
// v_wmma_f32_16x16x32_bf16 (wave32 WMMA). bf16 inputs, f32 accumulate.
// GEMM W tiles staged to LDS with GLOBAL_LOAD_ASYNC_TO_LDS_B128 when the
// toolchain exposes the builtin (ASYNCcnt-tracked), else sync staging.
// ---------------------------------------------------------------------------

typedef __bf16 bf16;
typedef __attribute__((ext_vector_type(16))) __bf16 v16bf;
typedef __attribute__((ext_vector_type(8)))  float  v8f;

// Model constants (match reference)
constexpr int cB  = 2;
constexpr int cT  = 1024;
constexpr int cH  = 6;
constexpr int cD  = 768;
constexpr int cHD = 128;
constexpr int cDF = 3072;
constexpr int cL  = 12;
constexpr int cV  = 50304;
constexpr int cBT = cB * cT;              // 2048 tokens

#if defined(__gfx1250__) && \
    __has_builtin(__builtin_amdgcn_global_load_async_to_lds_b128) && \
    __has_builtin(__builtin_amdgcn_s_wait_asynccnt)
#define USE_ASYNC_LDS 1
#else
#define USE_ASYNC_LDS 0
#endif

#if USE_ASYNC_LDS
// Builtin signature (from hipcc diagnostic): param0 is a *non-const* pointer
// to a 16-byte int vector in the global address space (AS1, printed as
// "__device__"); param1 is the LDS-side pointer (AS3). Build both via integer
// casts: LDS generic addresses carry the LDS byte offset in the low 32 bits.
typedef int v4i_async __attribute__((ext_vector_type(4)));
__device__ __forceinline__ void async_copy16(const void* g, void* l) {
  typedef __attribute__((address_space(1))) v4i_async* gp_t;
  typedef __attribute__((address_space(3))) v4i_async* lp_t;
  __builtin_amdgcn_global_load_async_to_lds_b128(
      (gp_t)(unsigned long long)(uintptr_t)g,
      (lp_t)(unsigned int)(uintptr_t)l, 0, 0);
}
#endif

// ---------------------------------------------------------------------------
// WMMA fragment helpers (A/B use contiguous k = 16*(lane>=16) + e, wave32)
// ---------------------------------------------------------------------------

// A fragment: 16x32 bf16, A row-major (lda elements). lane<16 holds row M=lane,
// K = koff+0..15, koff = 16*(lane>=16).
__device__ __forceinline__ v16bf load_a_bf(const bf16* A, int lda, int row0,
                                           int k0, int lane) {
  int row  = row0 + (lane & 15);
  int koff = (lane >> 4) << 4;
  const bf16* p = A + (long)row * lda + k0 + koff;
  v16bf a;
#pragma unroll
  for (int e = 0; e < 16; ++e) a[e] = p[e];
  return a;
}

// B fragment from an f32 LDS tile, converted to bf16 at read time.
// !TRANSB layout: wt[kk*64 + nn] (K-major).  TRANSB layout: wt[nn*32 + kk].
template <bool TRANSB>
__device__ __forceinline__ v16bf load_b_f32lds(const float* Wt, int nsub,
                                               int lane) {
  int col  = nsub + (lane & 15);
  int koff = (lane >> 4) << 4;
  v16bf b;
#pragma unroll
  for (int e = 0; e < 16; ++e) {
    float f = TRANSB ? Wt[col * 32 + koff + e] : Wt[(koff + e) * 64 + col];
    b[e] = (bf16)f;
  }
  return b;
}

// B fragment where the logical B[k][n] is stored transposed in global memory
// as Bm[n][k] (row stride ld). Used for Q.K^T (keys row-major over hd).
__device__ __forceinline__ v16bf load_bT_global(const bf16* Bm, int ld, int n0,
                                                int k0, int lane) {
  int row  = n0 + (lane & 15);
  int koff = (lane >> 4) << 4;
  const bf16* p = Bm + (long)row * ld + k0 + koff;
  v16bf b;
#pragma unroll
  for (int e = 0; e < 16; ++e) b[e] = p[e];
  return b;
}

// B fragment from a row-major global B[k][n] (row stride ld). Used for att.V.
__device__ __forceinline__ v16bf load_b_rowmajor(const bf16* Bm, int ld, int n0,
                                                 int k0, int lane) {
  int col  = n0 + (lane & 15);
  int koff = (lane >> 4) << 4;
  v16bf b;
#pragma unroll
  for (int e = 0; e < 16; ++e) b[e] = Bm[(long)(k0 + koff + e) * ld + col];
  return b;
}

__device__ __forceinline__ v8f wmma_bf16(v16bf a, v16bf b, v8f c) {
  return __builtin_amdgcn_wmma_f32_16x16x32_bf16(
      /*neg_a=*/false, a, /*neg_b=*/false, b,
      /*c_mod=*/(short)0, c, /*reuse_a=*/false, /*reuse_b=*/false);
}

// ---------------------------------------------------------------------------
// WMMA GEMM: C[M,N] = epi(A[M,K](bf16) @ W[K,N](f32) [+R])
// Block = 128 threads (4 waves). Block tile: 64 M x 64 N; wave w owns rows
// [m0+16w, m0+16w+16) and all 64 N cols (4 accumulators -> 4 WMMA / K-step).
// EPI: 0 none, 1 +R residual, 2 relu^2, 3 softcap tanh.
// TRANSB: W stored as [N,K] (lm_head).
// ---------------------------------------------------------------------------
template <int EPI, bool TRANSB>
__global__ __launch_bounds__(128) void gemm_wmma(
    const bf16* __restrict__ A, const float* __restrict__ W,
    const float* __restrict__ R, float* __restrict__ C,
    bf16* __restrict__ Cbf, int M, int N, int K) {
  __shared__ float wt[32 * 64];  // 8KB f32 W tile

  const int tid  = threadIdx.x;
  const int lane = tid & 31;
  const int wv   = tid >> 5;
  const int n0   = blockIdx.x * 64;
  const int m0   = blockIdx.y * 64;

  v8f acc[4] = {{}, {}, {}, {}};
  for (int kb = 0; kb < K; kb += 32) {
    // ---- stage 32(K) x 64(N) f32 W tile into LDS ----
#if USE_ASYNC_LDS
#pragma unroll
    for (int c = 0; c < 4; ++c) {
      const int chunk = tid + c * 128;  // 512 x 16B chunks
      const float* g;
      if (TRANSB) {
        int nn = chunk >> 3, kk4 = (chunk & 7) << 2;
        g = W + (long)(n0 + nn) * K + kb + kk4;
      } else {
        int kk = chunk >> 4, nn4 = (chunk & 15) << 2;
        g = W + (long)(kb + kk) * N + n0 + nn4;
      }
      async_copy16(g, wt + chunk * 4);
    }
    __builtin_amdgcn_s_wait_asynccnt(0);
#else
    if (TRANSB) {
      for (int i = tid; i < 32 * 64; i += 128) {
        int kk = i & 31, nn = i >> 5;
        wt[nn * 32 + kk] = W[(long)(n0 + nn) * K + kb + kk];
      }
    } else {
      for (int i = tid; i < 32 * 64; i += 128) {
        int kk = i >> 6, nn = i & 63;
        wt[kk * 64 + nn] = W[(long)(kb + kk) * N + n0 + nn];
      }
    }
#endif
    __syncthreads();

    // prefetch next K tile of W into cache
    if (kb + 32 < K) {
      const float* pf = TRANSB ? W + (long)(n0 + (tid >> 1)) * K + kb + 32
                               : W + (long)(kb + 32 + (tid >> 2)) * N + n0;
      __builtin_prefetch(pf, 0, 3);
    }

    v16bf a = load_a_bf(A, K, m0 + wv * 16, kb, lane);
#pragma unroll
    for (int nt = 0; nt < 4; ++nt) {
      v16bf b = load_b_f32lds<TRANSB>(wt, nt * 16, lane);
      acc[nt] = wmma_bf16(a, b, acc[nt]);
    }
    __syncthreads();
  }

  // C/D layout: VGPR r -> row 8*(lane>=16) + r, col = lane&15.
  const int row = m0 + wv * 16 + ((lane >> 4) << 3);
#pragma unroll
  for (int nt = 0; nt < 4; ++nt) {
    const int col = n0 + nt * 16 + (lane & 15);
#pragma unroll
    for (int r = 0; r < 8; ++r) {
      long idx = (long)(row + r) * N + col;
      float v = acc[nt][r];
      if (EPI == 1) v += R[idx];
      if (EPI == 2) { float t = fmaxf(v, 0.0f); v = t * t; }
      if (EPI == 3) v = 30.0f * tanhf(v * (1.0f / 30.0f));
      C[idx] = v;
      if (Cbf) Cbf[idx] = (bf16)v;
    }
  }
}

// ---------------------------------------------------------------------------
// Causal attention, one wave per (b, h, 16-query tile). Q/K/V are bf16 in
// [B,H,T,HD] layout. Scores staged in 64KB dynamic LDS; fully-masked key
// tiles are skipped. Output written bf16 in [b,t, h*HD+d] for the Wo GEMM.
// ---------------------------------------------------------------------------
__global__ __launch_bounds__(32) void attn_wmma(const bf16* __restrict__ Qh,
                                                const bf16* __restrict__ Kh,
                                                const bf16* __restrict__ Vh,
                                                bf16* __restrict__ Ybf) {
  extern __shared__ float sc[];  // 16 x 1024 f32 scores/probs
  const int lane = threadIdx.x;
  const int qt = blockIdx.x & 63;   // T/16 = 64 query tiles
  const int bh = blockIdx.x >> 6;   // 0..B*H-1
  const int b = bh / cH, h = bh % cH;
  const bf16* Qp = Qh + (long)bh * cT * cHD;
  const bf16* Kp = Kh + (long)bh * cT * cHD;
  const bf16* Vp = Vh + (long)bh * cT * cHD;
  const int rbase = (lane >> 4) << 3;

  // Q fragments for the whole HD=128 (4 chunks of K=32)
  v16bf aq[4];
#pragma unroll
  for (int c = 0; c < 4; ++c) aq[c] = load_a_bf(Qp, cHD, qt * 16, c * 32, lane);

  const float scale = 0.0883883476f;  // 1/sqrt(128)

  // ---- scores: S = Q K^T (only kt <= qt survives the causal mask) ----
  for (int kt = 0; kt <= qt; ++kt) {
    v8f s = {};
#pragma unroll
    for (int c = 0; c < 4; ++c) {
      v16bf bk = load_bT_global(Kp, cHD, kt * 16, c * 32, lane);
      s = wmma_bf16(aq[c], bk, s);
    }
    const int col = kt * 16 + (lane & 15);
#pragma unroll
    for (int r = 0; r < 8; ++r) {
      int rowq = qt * 16 + rbase + r;
      sc[(rbase + r) * cT + col] = (col <= rowq) ? s[r] * scale : -1e30f;
    }
  }
  __syncthreads();

  // ---- softmax over [0, kmax); zero-pad to a 32-multiple for att@V ----
  const int kmax = (qt + 1) * 16;
  const int kpad = (kmax + 31) & ~31;
  if (lane < 16) {
    float* rowp = sc + lane * cT;
    float mx = -1e30f;
    for (int c = 0; c < kmax; ++c) mx = fmaxf(mx, rowp[c]);
    float sum = 0.0f;
    for (int c = 0; c < kmax; ++c) {
      float e = __expf(rowp[c] - mx);
      rowp[c] = e;
      sum += e;
    }
    float inv = 1.0f / sum;
    for (int c = 0; c < kmax; ++c) rowp[c] *= inv;
    for (int c = kmax; c < kpad; ++c) rowp[c] = 0.0f;
  }
  __syncthreads();

  // ---- y = att @ V : 8 output tiles of 16 hd-cols, K-loop over keys ----
  const long gtrow = (long)b * cT + qt * 16;
#pragma unroll
  for (int nt = 0; nt < 8; ++nt) {
    v8f acc = {};
    for (int kk = 0; kk < kpad; kk += 32) {
      v16bf a;
      const int arow = lane & 15;
      const int koff = (lane >> 4) << 4;
      const float* p = sc + arow * cT + kk + koff;
#pragma unroll
      for (int e = 0; e < 16; ++e) {
        a[e] = (bf16)p[e];
      }
      v16bf bv = load_b_rowmajor(Vp, cHD, nt * 16, kk, lane);
      acc = wmma_bf16(a, bv, acc);
    }
#pragma unroll
    for (int r = 0; r < 8; ++r) {
      long grow = gtrow + rbase + r;
      Ybf[grow * cD + h * cHD + nt * 16 + (lane & 15)] = (bf16)acc[r];
    }
  }
}

// ---------------------------------------------------------------------------
// Elementwise / norm kernels
// ---------------------------------------------------------------------------

// x = x0 = rmsnorm(wte[idx[t]]); one block (256 thr) per token, 3 elems/thr.
__global__ __launch_bounds__(256) void embed_rmsnorm(
    const int* __restrict__ idx, const float* __restrict__ wte,
    float* __restrict__ x, float* __restrict__ x0) {
  __shared__ float red[256];
  const int t = blockIdx.x;
  const float* src = wte + (long)idx[t] * cD;
  float v[3];
  float ss = 0.0f;
#pragma unroll
  for (int j = 0; j < 3; ++j) {
    v[j] = src[threadIdx.x + 256 * j];
    ss += v[j] * v[j];
  }
  red[threadIdx.x] = ss;
  __syncthreads();
  for (int s = 128; s > 0; s >>= 1) {
    if (threadIdx.x < s) red[threadIdx.x] += red[threadIdx.x + s];
    __syncthreads();
  }
  const float inv = rsqrtf(red[0] / (float)cD + 1e-6f);
  const long base = (long)t * cD;
#pragma unroll
  for (int j = 0; j < 3; ++j) {
    float o = v[j] * inv;
    x[base + threadIdx.x + 256 * j] = o;
    x0[base + threadIdx.x + 256 * j] = o;
  }
}

// xi = l0*x + l1*x0 ; xn = (bf16)rmsnorm(xi)
__global__ __launch_bounds__(256) void mix_rmsnorm(
    const float* __restrict__ x, const float* __restrict__ x0,
    const float* __restrict__ lambdas, int layer, float* __restrict__ xi,
    bf16* __restrict__ xn) {
  __shared__ float red[256];
  const float l0 = lambdas[2 * layer], l1 = lambdas[2 * layer + 1];
  const long base = (long)blockIdx.x * cD;
  float v[3];
  float ss = 0.0f;
#pragma unroll
  for (int j = 0; j < 3; ++j) {
    int c = threadIdx.x + 256 * j;
    float val = l0 * x[base + c] + l1 * x0[base + c];
    v[j] = val;
    ss += val * val;
    xi[base + c] = val;
  }
  red[threadIdx.x] = ss;
  __syncthreads();
  for (int s = 128; s > 0; s >>= 1) {
    if (threadIdx.x < s) red[threadIdx.x] += red[threadIdx.x + s];
    __syncthreads();
  }
  const float inv = rsqrtf(red[0] / (float)cD + 1e-6f);
#pragma unroll
  for (int j = 0; j < 3; ++j) {
    int c = threadIdx.x + 256 * j;
    xn[base + c] = (bf16)(v[j] * inv);
  }
}

// xn = (bf16)rmsnorm(x)
__global__ __launch_bounds__(256) void rmsnorm_to_bf(const float* __restrict__ x,
                                                     bf16* __restrict__ xn) {
  __shared__ float red[256];
  const long base = (long)blockIdx.x * cD;
  float v[3];
  float ss = 0.0f;
#pragma unroll
  for (int j = 0; j < 3; ++j) {
    v[j] = x[base + threadIdx.x + 256 * j];
    ss += v[j] * v[j];
  }
  red[threadIdx.x] = ss;
  __syncthreads();
  for (int s = 128; s > 0; s >>= 1) {
    if (threadIdx.x < s) red[threadIdx.x] += red[threadIdx.x + s];
    __syncthreads();
  }
  const float inv = rsqrtf(red[0] / (float)cD + 1e-6f);
#pragma unroll
  for (int j = 0; j < 3; ++j)
    xn[base + threadIdx.x + 256 * j] = (bf16)(v[j] * inv);
}

// Per (token, head): q,k <- rotary(rmsnorm(.)), v <- lamb mix (+ v1 save at
// layer 0). Writes bf16 in [B,H,T,HD] for the attention kernel.
__global__ __launch_bounds__(128) void qkv_prep(
    const float* __restrict__ q, const float* __restrict__ k,
    const float* __restrict__ v, float* __restrict__ v1,
    const float* __restrict__ lamb, int layer, bf16* __restrict__ qh,
    bf16* __restrict__ kh, bf16* __restrict__ vh) {
  __shared__ float red[128];
  __shared__ float buf[128];
  const int bt = blockIdx.x / cH;
  const int h  = blockIdx.x % cH;
  const int d  = threadIdx.x;  // 0..127
  const int pos = bt & (cT - 1);
  const int b   = bt >> 10;
  const long ibase = (long)bt * cD + h * cHD;
  const long obase = ((long)(b * cH + h) * cT + pos) * cHD;
  const float l0 = lamb[2 * layer], l1 = lamb[2 * layer + 1];

  // rotary coefficients for this element
  const int i2 = d & 63;
  const float invf = __expf(-9.210340372f * ((float)i2 / 64.0f));  // 10000^(-i/64)
  const float ang = invf * (float)pos;
  const float ca = __cosf(ang), sa = __sinf(ang);

  // ---- Q: rmsnorm over HD then rotary ----
  float qv = q[ibase + d];
  red[d] = qv * qv;
  __syncthreads();
  for (int s = 64; s > 0; s >>= 1) {
    if (d < s) red[d] += red[d + s];
    __syncthreads();
  }
  float inv = rsqrtf(red[0] / (float)cHD + 1e-6f);
  buf[d] = qv * inv;
  __syncthreads();
  float qo = (d < 64) ? (buf[d] * ca + buf[d + 64] * sa)
                      : (-buf[d - 64] * sa + buf[d] * ca);
  qh[obase + d] = (bf16)qo;
  __syncthreads();

  // ---- K ----
  float kv = k[ibase + d];
  red[d] = kv * kv;
  __syncthreads();
  for (int s = 64; s > 0; s >>= 1) {
    if (d < s) red[d] += red[d + s];
    __syncthreads();
  }
  float invk = rsqrtf(red[0] / (float)cHD + 1e-6f);
  buf[d] = kv * invk;
  __syncthreads();
  float ko = (d < 64) ? (buf[d] * ca + buf[d + 64] * sa)
                      : (-buf[d - 64] * sa + buf[d] * ca);
  kh[obase + d] = (bf16)ko;

  // ---- V: save v1 at layer 0, mix ----
  float vv = v[ibase + d];
  if (layer == 0) v1[ibase + d] = vv;
  float vm = l0 * vv + l1 * v1[ibase + d];
  vh[obase + d] = (bf16)vm;
}

__global__ void copy_f32(const float* __restrict__ src, float* __restrict__ dst,
                         long n) {
  long i = (long)blockIdx.x * blockDim.x + threadIdx.x;
  if (i < n) dst[i] = src[i];
}

// ---------------------------------------------------------------------------
// Host launcher
// ---------------------------------------------------------------------------
extern "C" void kernel_launch(void* const* d_in, const int* in_sizes, int n_in,
                              void* d_out, int out_size, void* d_ws,
                              size_t ws_size, hipStream_t stream) {
  const int*   idx     = (const int*)d_in[0];
  const float* wte     = (const float*)d_in[1];
  const float* lambdas = (const float*)d_in[2];
  const float* lamb    = (const float*)d_in[3];
  const float* Wq      = (const float*)d_in[4];
  const float* Wk      = (const float*)d_in[5];
  const float* Wv      = (const float*)d_in[6];
  const float* Wo      = (const float*)d_in[7];
  const float* Wfc     = (const float*)d_in[8];
  const float* Wproj   = (const float*)d_in[9];
  const float* lmh     = (const float*)d_in[10];
  float* out = (float*)d_out;

  // carve workspace
  char* ws = (char*)d_ws;
  auto alloc = [&](size_t bytes) -> char* {
    char* p = ws;
    ws += (bytes + 255) & ~(size_t)255;
    return p;
  };
  const size_t fBTD = (size_t)cBT * cD * sizeof(float);
  float* x   = (float*)alloc(fBTD);
  float* x0  = (float*)alloc(fBTD);
  float* xi  = (float*)alloc(fBTD);
  float* q   = (float*)alloc(fBTD);
  float* kb_ = (float*)alloc(fBTD);
  float* v   = (float*)alloc(fBTD);
  float* v1  = (float*)alloc(fBTD);
  float* fc  = (float*)alloc((size_t)cBT * cDF * sizeof(float));
  bf16* xn_bf = (bf16*)alloc((size_t)cBT * cD * sizeof(bf16));
  bf16* ybf   = (bf16*)alloc((size_t)cBT * cD * sizeof(bf16));
  bf16* qh    = (bf16*)alloc((size_t)cBT * cD * sizeof(bf16));
  bf16* kh    = (bf16*)alloc((size_t)cBT * cD * sizeof(bf16));
  bf16* vh    = (bf16*)alloc((size_t)cBT * cD * sizeof(bf16));
  bf16* fcbf  = (bf16*)alloc((size_t)cBT * cDF * sizeof(bf16));

  // GEMM block tile is 64x64
  const dim3 gD(cD / 64, cBT / 64);     // N=768
  const dim3 gDF(cDF / 64, cBT / 64);   // N=3072
  const dim3 gV(cV / 64, cBT / 64);     // N=50304
  const dim3 blk128(128);

  // x = x0 = rmsnorm(wte[idx])
  embed_rmsnorm<<<cBT, 256, 0, stream>>>(idx, wte, x, x0);

  for (int i = 0; i < cL; ++i) {
    const float* wq = Wq + (size_t)i * cD * cD;
    const float* wk = Wk + (size_t)i * cD * cD;
    const float* wv = Wv + (size_t)i * cD * cD;
    const float* wo = Wo + (size_t)i * cD * cD;
    const float* wf = Wfc + (size_t)i * cD * cDF;
    const float* wp = Wproj + (size_t)i * cDF * cD;

    // xi = l0*x + l1*x0 ; xn = rmsnorm(xi) (bf16)
    mix_rmsnorm<<<cBT, 256, 0, stream>>>(x, x0, lambdas, i, xi, xn_bf);

    // q,k,v projections
    gemm_wmma<0, false><<<gD, blk128, 0, stream>>>(xn_bf, wq, nullptr, q,
                                                   nullptr, cBT, cD, cD);
    gemm_wmma<0, false><<<gD, blk128, 0, stream>>>(xn_bf, wk, nullptr, kb_,
                                                   nullptr, cBT, cD, cD);
    gemm_wmma<0, false><<<gD, blk128, 0, stream>>>(xn_bf, wv, nullptr, v,
                                                   nullptr, cBT, cD, cD);

    // head rmsnorm + rotary + v-mix -> bf16 [B,H,T,HD]
    qkv_prep<<<cBT * cH, 128, 0, stream>>>(q, kb_, v, v1, lamb, i, qh, kh, vh);

    // attention (one wave per b,h,qtile), 64KB dynamic LDS
    attn_wmma<<<cB * cH * (cT / 16), 32, 16 * cT * sizeof(float), stream>>>(
        qh, kh, vh, ybf);

    // x = xi + y @ Wo
    gemm_wmma<1, false><<<gD, blk128, 0, stream>>>(ybf, wo, xi, x, nullptr,
                                                   cBT, cD, cD);

    // MLP: x = x + relu(rmsnorm(x) @ Wfc)^2 @ Wproj
    rmsnorm_to_bf<<<cBT, 256, 0, stream>>>(x, xn_bf);
    gemm_wmma<2, false><<<gDF, blk128, 0, stream>>>(xn_bf, wf, nullptr, fc,
                                                    fcbf, cBT, cDF, cD);
    gemm_wmma<1, false><<<gD, blk128, 0, stream>>>(fcbf, wp, x, x, nullptr,
                                                   cBT, cD, cDF);
  }

  // logits = softcap(rmsnorm(x) @ lm_head^T), h = x
  rmsnorm_to_bf<<<cBT, 256, 0, stream>>>(x, xn_bf);
  gemm_wmma<3, true><<<gV, blk128, 0, stream>>>(xn_bf, lmh, nullptr, out,
                                                nullptr, cBT, cV, cD);

  const long hN = (long)cBT * cD;
  copy_f32<<<(int)((hN + 255) / 256), 256, 0, stream>>>(
      x, out + (size_t)cBT * cV, hN);
}
